// LateMultiInteractiveRetriever_10754598109467
// MI455X (gfx1250) — matrix operand order repaired and checked by reading, not compiled, not run on hardware
//
#include <hip/hip_runtime.h>

// ---------------------------------------------------------------------------
// CDNA5 (gfx1250) implementation: all GEMMs via v_wmma_f32_16x16x32_bf16.
// Maxsim kernel stages K-rows in LDS (ds_load_b128 A-operand feed), Q
// fragments hoisted to registers.
// ---------------------------------------------------------------------------

typedef __bf16 bf16_t;
typedef __attribute__((ext_vector_type(16))) __bf16 v16bf;
typedef __attribute__((ext_vector_type(8)))  __bf16 v8bf;
typedef __attribute__((ext_vector_type(8)))  float  v8f;

#define DMODEL 256
#define NCAND  2000
#define NQ     1024           // 2 * 512 query rows
#define JSW    32
#define JPHO   24

// -------------------- WMMA fragment loaders --------------------------------
// A operand (16xK=32 tile of a row-major [*, 256] matrix):
// lane L -> row rowBase+(L&15); elements 0..7 at kk+8*(L>>4), 8..15 at +16.
__device__ __forceinline__ v16bf load_a_frag_bf16(const bf16_t* __restrict__ base,
                                                  int rowBase, int kk, int lane) {
  const int m  = rowBase + (lane & 15);
  const int kb = kk + ((lane >> 4) << 3);
  const bf16_t* p = base + m * DMODEL + kb;
  v8bf lo = *(const v8bf*)(p);
  v8bf hi = *(const v8bf*)(p + 16);
  v16bf a;
#pragma unroll
  for (int i = 0; i < 8; ++i) { a[i] = lo[i]; a[i + 8] = hi[i]; }
  return a;
}

// Same pattern but from an LDS-resident tile buffer (rows of 256 bf16).
__device__ __forceinline__ v16bf load_a_frag_lds(const bf16_t* buf,
                                                 int rowBase, int kk, int lane) {
  const int m  = rowBase + (lane & 15);
  const int kb = kk + ((lane >> 4) << 3);
  const bf16_t* p = buf + m * DMODEL + kb;
  v8bf lo = *(const v8bf*)(p);
  v8bf hi = *(const v8bf*)(p + 16);
  v16bf a;
#pragma unroll
  for (int i = 0; i < 8; ++i) { a[i] = lo[i]; a[i + 8] = hi[i]; }
  return a;
}

__device__ __forceinline__ v16bf load_a_frag_f32(const float* __restrict__ base,
                                                 int rowBase, int kk, int lane) {
  const int m  = rowBase + (lane & 15);
  const int kb = kk + ((lane >> 4) << 3);
  const float* p = base + m * DMODEL + kb;
  v16bf a;
#pragma unroll
  for (int i = 0; i < 8; ++i) { a[i] = (bf16_t)p[i]; a[i + 8] = (bf16_t)p[i + 16]; }
  return a;
}

// B operand (K=32 x 16 tile of X^T where X is row-major [*, 256]):
// lane L -> column colBase+(L&15) = row of X; elements 0..7 at kk+8*(L>>4), 8..15 at +16.
__device__ __forceinline__ v16bf load_b_frag(const bf16_t* __restrict__ base,
                                             int colBase, int kk, int lane) {
  const int n  = colBase + (lane & 15);
  const int kb = kk + ((lane >> 4) << 3);
  const bf16_t* p = base + n * DMODEL + kb;
  v8bf lo = *(const v8bf*)(p);
  v8bf hi = *(const v8bf*)(p + 16);
  v16bf b;
#pragma unroll
  for (int i = 0; i < 8; ++i) { b[i] = lo[i]; b[i + 8] = hi[i]; }
  return b;
}

// -------------------- f32 -> bf16 conversion (weights) ---------------------
__global__ __launch_bounds__(256)
void f32_to_bf16_kernel(const float* __restrict__ src, bf16_t* __restrict__ dst, int n) {
  int i = blockIdx.x * 256 + threadIdx.x;
  if (i < n) dst[i] = (bf16_t)src[i];
}

// -------------------- Linear layer:  out = A @ W^T + b  (bf16 out) ---------
// A: [M,256] (f32 or bf16), W: [256,256] bf16 row-major (torch Linear weight),
// bias: [256] f32.  Block = 256 thr = 8 waves; wave handles one 16-col tile.
template <bool A_IS_F32>
__global__ __launch_bounds__(256)
void gemm_nt_bias_kernel(const void* __restrict__ Aptr,
                         const bf16_t* __restrict__ W,
                         const float* __restrict__ bias,
                         bf16_t* __restrict__ out, int M) {
  const int lane  = threadIdx.x & 31;
  const int wave  = threadIdx.x >> 5;
  const int mBase = blockIdx.x * 16;
  const int nBase = (blockIdx.y * 8 + wave) * 16;
  if (mBase >= M) return;   // wave-uniform

  v8f acc = {};
#pragma unroll
  for (int ks = 0; ks < 8; ++ks) {
    const int kk = ks * 32;
    v16bf a = A_IS_F32 ? load_a_frag_f32((const float*)Aptr, mBase, kk, lane)
                       : load_a_frag_bf16((const bf16_t*)Aptr, mBase, kk, lane);
    v16bf b = load_b_frag(W, nBase, kk, lane);
    acc = __builtin_amdgcn_wmma_f32_16x16x32_bf16(false, a, false, b,
                                                  (short)0, acc, false, false);
  }
  const int col  = nBase + (lane & 15);
  const float bv = bias[col];
  const int rOff = (lane >> 4) << 3;
#pragma unroll
  for (int r = 0; r < 8; ++r) {
    const int row = mBase + r + rOff;
    out[row * DMODEL + col] = (bf16_t)(acc[r] + bv);
  }
}

// -------------------- maxsim: lane-max of one 16x16 score tile -------------
// A = 16 context-token rows (LDS), B = 16 query columns (register-hoisted).
// Returns per-lane max over the 8 token rows this lane holds
// (lane = query column N=lane&15, rows M = r + 8*(lane>>4)).
__device__ __forceinline__ float tile_rowmax(const bf16_t* kbuf, int rowBase,
                                             const v16bf (&qf)[8], int lane) {
  v8f acc = {};
#pragma unroll
  for (int ks = 0; ks < 8; ++ks) {
    v16bf a = load_a_frag_lds(kbuf, rowBase, ks * 32, lane);
    acc = __builtin_amdgcn_wmma_f32_16x16x32_bf16(false, a, false, qf[ks],
                                                  (short)0, acc, false, false);
  }
  float m = acc[0];
#pragma unroll
  for (int r = 1; r < 8; ++r) m = fmaxf(m, acc[r]);
  return m;
}

// Block = 8 waves x 1 candidate pair; wave w covers query tile
// blockIdx.x*8 + w. K rows of the pair (64 sw + 48 pho = 56 KB) are staged
// into LDS once and reused by all 8 waves.
__global__ __launch_bounds__(256)
void maxsim_kernel(const bf16_t* __restrict__ qsw, const bf16_t* __restrict__ qpho,
                   const bf16_t* __restrict__ ksw, const bf16_t* __restrict__ kpho,
                   float* __restrict__ logits) {
  __shared__ __align__(16) bf16_t kbuf[(64 + 48) * DMODEL];   // 57,344 B

  const int lane  = threadIdx.x & 31;
  const int wave  = threadIdx.x >> 5;
  const int qBase = (blockIdx.x * 8 + wave) * 16;
  const int cpair = blockIdx.y;                 // 0..999
  const int c0    = cpair * 2;
  const float NEG = -3.0e38f;

  // ---- stage K rows: sw pair rows are contiguous [64*cpair, +64) ----------
  {
    const uint4* gsw = (const uint4*)(ksw + (size_t)cpair * 64 * DMODEL);
    uint4* lsw = (uint4*)kbuf;
    for (int i = threadIdx.x; i < 64 * DMODEL / 8; i += 256) lsw[i] = gsw[i];
    const uint4* gph = (const uint4*)(kpho + (size_t)cpair * 48 * DMODEL);
    uint4* lph = (uint4*)(kbuf + 64 * DMODEL);
    for (int i = threadIdx.x; i < 48 * DMODEL / 8; i += 256) lph[i] = gph[i];
  }
  __syncthreads();

  // ---- sw phase: hoist q_sw fragments, 32 tokens/candidate = 2 tiles ------
  float sw0, sw1;
  {
    v16bf qf[8];
#pragma unroll
    for (int ks = 0; ks < 8; ++ks) qf[ks] = load_b_frag(qsw, qBase, ks * 32, lane);

    float m0 = tile_rowmax(kbuf, 0,  qf, lane);
    float m1 = tile_rowmax(kbuf, 16, qf, lane);
    float v  = fmaxf(m0, m1);
    v = fmaxf(v, __shfl_xor(v, 16, 32));
    sw0 = v;
    m0 = tile_rowmax(kbuf, 32, qf, lane);
    m1 = tile_rowmax(kbuf, 48, qf, lane);
    v  = fmaxf(m0, m1);
    v = fmaxf(v, __shfl_xor(v, 16, 32));
    sw1 = v;
  }

  // ---- pho phase: 24 tokens/candidate -> pair = 48 rows = 3 tiles ---------
  // middle tile: M=0..7 (lanes 0-15) -> c0 tokens 16..23,
  //              M=8..15 (lanes 16-31) -> c1 tokens 0..7.
  float v0, v1;
  {
    v16bf qf[8];
#pragma unroll
    for (int ks = 0; ks < 8; ++ks) qf[ks] = load_b_frag(qpho, qBase, ks * 32, lane);

    const int pBase = 64 * 16 / 16;  // LDS row 64
    float p0 = tile_rowmax(kbuf, 64,      qf, lane);
    float p1 = tile_rowmax(kbuf, 64 + 16, qf, lane);
    float p2 = tile_rowmax(kbuf, 64 + 32, qf, lane);
    (void)pBase;

    v0 = fmaxf(p0, (lane < 16) ? p1 : NEG);
    v0 = fmaxf(v0, __shfl_xor(v0, 16, 32));
    v1 = fmaxf(p2, (lane >= 16) ? p1 : NEG);
    v1 = fmaxf(v1, __shfl_xor(v1, 16, 32));
  }

  if (lane < 16) {
    const int t = qBase + lane;
    logits[t * NCAND + c0]     = sw0 + v0;
    logits[t * NCAND + c0 + 1] = sw1 + v1;
  }
}

// -------------------- row softmax over 2000 candidates ---------------------
__global__ __launch_bounds__(256)
void softmax_rows_kernel(const float* __restrict__ logits, float* __restrict__ out) {
  __shared__ float red[256];
  const int t = blockIdx.x;
  const float* row = logits + (long)t * NCAND;

  float m = -3.4e38f;
  for (int c = threadIdx.x; c < NCAND; c += 256) m = fmaxf(m, row[c]);
  red[threadIdx.x] = m; __syncthreads();
  for (int s = 128; s > 0; s >>= 1) {
    if (threadIdx.x < s) red[threadIdx.x] = fmaxf(red[threadIdx.x], red[threadIdx.x + s]);
    __syncthreads();
  }
  m = red[0]; __syncthreads();

  float sum = 0.f;
  for (int c = threadIdx.x; c < NCAND; c += 256) {
    float e = __expf(row[c] - m);
    out[(long)t * NCAND + c] = e;
    sum += e;
  }
  red[threadIdx.x] = sum; __syncthreads();
  for (int s = 128; s > 0; s >>= 1) {
    if (threadIdx.x < s) red[threadIdx.x] += red[threadIdx.x + s];
    __syncthreads();
  }
  const float inv = 1.0f / red[0];
  for (int c = threadIdx.x; c < NCAND; c += 256) out[(long)t * NCAND + c] *= inv;
}

// ---------------------------------------------------------------------------
extern "C" void kernel_launch(void* const* d_in, const int* in_sizes, int n_in,
                              void* d_out, int out_size, void* d_ws, size_t ws_size,
                              hipStream_t stream) {
  (void)in_sizes; (void)n_in; (void)out_size; (void)ws_size;

  const float* model_embed   = (const float*)d_in[0];   // [2,512,256]
  const float* context_embed = (const float*)d_in[1];   // [2000,32,256]
  const float* xphone_embed  = (const float*)d_in[2];   // [2000,24,256]
  const float* Wq = (const float*)d_in[3];  const float* bq = (const float*)d_in[4];
  const float* W1 = (const float*)d_in[5];  const float* b1 = (const float*)d_in[6];
  const float* W2 = (const float*)d_in[7];  const float* b2 = (const float*)d_in[8];
  const float* W3 = (const float*)d_in[9];  const float* b3 = (const float*)d_in[10];
  const float* W4 = (const float*)d_in[11]; const float* b4 = (const float*)d_in[12];
  float* out = (float*)d_out;

  // ---- workspace layout (bytes) ----
  char* ws = (char*)d_ws;
  const size_t WSZ = (size_t)DMODEL * DMODEL * 2;         // 128 KB per weight
  bf16_t* wq_bf   = (bf16_t*)(ws + 0 * WSZ);
  bf16_t* w1_bf   = (bf16_t*)(ws + 1 * WSZ);
  bf16_t* w2_bf   = (bf16_t*)(ws + 2 * WSZ);
  bf16_t* w3_bf   = (bf16_t*)(ws + 3 * WSZ);
  bf16_t* w4_bf   = (bf16_t*)(ws + 4 * WSZ);
  size_t off = 5 * WSZ;
  bf16_t* query_bf = (bf16_t*)(ws + off); off += (size_t)NQ * DMODEL * 2;
  bf16_t* qsw_bf   = (bf16_t*)(ws + off); off += (size_t)NQ * DMODEL * 2;
  bf16_t* qpho_bf  = (bf16_t*)(ws + off); off += (size_t)NQ * DMODEL * 2;
  float*  logits   = (float*)(ws + off);  off += (size_t)NQ * NCAND * 4;
  bf16_t* ksw_bf   = (bf16_t*)(ws + off); off += (size_t)NCAND * JSW * DMODEL * 2;
  bf16_t* kpho_bf  = (bf16_t*)(ws + off); off += (size_t)NCAND * JPHO * DMODEL * 2;

  const int WN = DMODEL * DMODEL;   // 65536
  dim3 blk(256);

  // 1) weights -> bf16
  f32_to_bf16_kernel<<<dim3(WN / 256), blk, 0, stream>>>(Wq, wq_bf, WN);
  f32_to_bf16_kernel<<<dim3(WN / 256), blk, 0, stream>>>(W1, w1_bf, WN);
  f32_to_bf16_kernel<<<dim3(WN / 256), blk, 0, stream>>>(W2, w2_bf, WN);
  f32_to_bf16_kernel<<<dim3(WN / 256), blk, 0, stream>>>(W3, w3_bf, WN);
  f32_to_bf16_kernel<<<dim3(WN / 256), blk, 0, stream>>>(W4, w4_bf, WN);

  // 2) linear layers (all WMMA)
  gemm_nt_bias_kernel<true ><<<dim3(NQ / 16, 2), blk, 0, stream>>>(model_embed,   wq_bf, bq, query_bf, NQ);
  gemm_nt_bias_kernel<false><<<dim3(NQ / 16, 2), blk, 0, stream>>>(query_bf,      w1_bf, b1, qsw_bf,   NQ);
  gemm_nt_bias_kernel<false><<<dim3(NQ / 16, 2), blk, 0, stream>>>(query_bf,      w3_bf, b3, qpho_bf,  NQ);
  gemm_nt_bias_kernel<true ><<<dim3(NCAND * JSW  / 16, 2), blk, 0, stream>>>(context_embed, w2_bf, b2, ksw_bf,  NCAND * JSW);
  gemm_nt_bias_kernel<true ><<<dim3(NCAND * JPHO / 16, 2), blk, 0, stream>>>(xphone_embed,  w4_bf, b4, kpho_bf, NCAND * JPHO);

  // 3) late-interaction maxsim (WMMA, LDS-staged K) -> logits [1024, 2000]
  maxsim_kernel<<<dim3(NQ / 128, NCAND / 2), blk, 0, stream>>>(
      qsw_bf, qpho_bf, ksw_bf, kpho_bf, logits);

  // 4) softmax over candidates -> d_out
  softmax_rows_kernel<<<dim3(NQ), blk, 0, stream>>>(logits, out);
}
